// MnistDetector_59622736003951
// MI455X (gfx1250) — matrix operand order repaired and checked by reading, not compiled, not run on hardware
//
#include <hip/hip_runtime.h>
#include <hip/hip_bf16.h>

typedef __attribute__((ext_vector_type(16))) _Float16 v16h;
typedef __attribute__((ext_vector_type(8)))  float    v8f;

#define NBATCH 4
#define HP 22
#define WP 22
#define FEAT_HW (HP*WP)          // 484
#define NANCH (9*FEAT_HW)        // 4356
#define NREG 256
#define RXY 28

// ---------------- weight pre-pack into WMMA A-operand layout ----------------
// pw[((mB*Ksteps + s)*32 + lane)*16 + i] = f16(W[M, k]) with
//   M = mB*16 + lane%16 ; k = s*32 + (i<8 ? i : i+8) + 8*(lane>=16)
__global__ void pack_weights_kernel(const float* __restrict__ w, _Float16* __restrict__ pw,
                                    int Cout, int Ktot, int nM, int Ksteps)
{
  int idx = blockIdx.x * blockDim.x + threadIdx.x;
  int total = nM * Ksteps * 512;
  if (idx >= total) return;
  int i    = idx & 15;
  int lane = (idx >> 4) & 31;
  int s    = (idx >> 9) % Ksteps;
  int mB   = idx / (Ksteps * 512);
  int M    = mB * 16 + (lane & 15);
  int k    = s * 32 + (i < 8 ? i : i + 8) + ((lane >> 4) ? 8 : 0);
  float v  = 0.0f;
  if (M < Cout && k < Ktot) v = w[(size_t)M * Ktot + k];
  pw[idx] = (_Float16)v;
}

// ---------------- im2col pre-pack into WMMA B-operand layout ----------------
// pb[(((b*nN + nB)*Ksteps + s)*32 + lane)*16 + i] = f16(im2col[k, N]) with
//   N = nB*16 + lane%16 ; k = s*32 + (i<8 ? i : i+8) + 8*(lane>=16)
// Clamped addresses + select: unconditional loads, no divergent branches.
__global__ void pack_im2col_kernel(const float* __restrict__ in, _Float16* __restrict__ pb,
                                   int Cin, int Hin, int Win, int Hout, int Wout,
                                   int pad, int ks, int Ktot, int Ksteps, int nN)
{
  int idx = blockIdx.x * blockDim.x + threadIdx.x;
  int total = NBATCH * nN * Ksteps * 512;
  if (idx >= total) return;
  int i    = idx & 15;
  int lane = (idx >> 4) & 31;
  int s    = (idx >> 9) % Ksteps;
  int t2   = idx / (Ksteps * 512);
  int nB   = t2 % nN;
  int b    = t2 / nN;
  int HW   = Hout * Wout;
  int N    = nB * 16 + (lane & 15);
  int k    = s * 32 + (i < 8 ? i : i + 8) + ((lane >> 4) ? 8 : 0);

  int kc  = min(k, Ktot - 1);
  int ks2 = ks * ks;
  int ci  = kc / ks2;
  int t   = kc - ci * ks2;
  int kh  = t / ks;
  int kw  = t - kh * ks;
  int Nc  = min(N, HW - 1);
  int oy  = Nc / Wout;
  int ox  = Nc - oy * Wout;
  int y   = oy + kh - pad;
  int x   = ox + kw - pad;
  int yc  = min(max(y, 0), Hin - 1);
  int xc  = min(max(x, 0), Win - 1);
  float v = in[(((size_t)b * Cin + ci) * Hin + yc) * Win + xc];
  bool ok = (N < HW) & (k < Ktot) & (y >= 0) & (y < Hin) & (x >= 0) & (x < Win);
  pb[idx] = ok ? (_Float16)v : (_Float16)0.0f;
}

// ---------------- WMMA GEMM: D = A(Cout x K) * B(K x HW) + bias -------------
// One wave per 16x16 tile; inner loop = 2 contiguous 32B loads + 1 wmma.
__global__ __launch_bounds__(128) void gemm16_wmma_kernel(
    const _Float16* __restrict__ pA, const _Float16* __restrict__ pB,
    const float* __restrict__ bias, float* __restrict__ out,
    int Cout, int nM, int HW, int nN, int Ksteps, int relu, int nTiles)
{
  int wid = blockIdx.x * (blockDim.x >> 5) + (threadIdx.x >> 5);
  if (wid >= nTiles) return;                       // wave-uniform
  int mB = wid % nM;
  int t  = wid / nM;
  int nB = t % nN;
  int b  = t / nN;
  int lane = threadIdx.x & 31;

  const _Float16* aP = pA + ((size_t)mB * Ksteps * 32 + lane) * 16;
  const _Float16* bP = pB + (((size_t)b * nN + nB) * Ksteps * 32 + lane) * 16;

  v8f acc = {};
  #pragma unroll 2
  for (int s = 0; s < Ksteps; ++s) {
    v16h av = *(const v16h*)(aP + (size_t)s * 512);
    v16h bv = *(const v16h*)(bP + (size_t)s * 512);
    acc = __builtin_amdgcn_wmma_f32_16x16x32_f16(
        false, av, false, bv, (short)0, acc, false, false);
  }

  int row = lane & 15, hi = lane >> 4;
  int N = nB * 16 + row;
  if (N < HW) {
    float* outB = out + (size_t)b * Cout * HW;
    #pragma unroll
    for (int r = 0; r < 8; ++r) {
      int m = mB * 16 + r + (hi ? 8 : 0);          // C/D layout: VGPR r -> M
      if (m < Cout) {
        float v = acc[r] + bias[m];
        if (relu) v = fmaxf(v, 0.0f);
        outB[(size_t)m * HW + N] = v;
      }
    }
  }
}

// ---------------- 2x2 maxpool, stride 2, VALID ----------------
__global__ void maxpool2_kernel(const float* __restrict__ in, float* __restrict__ out,
                                int NC, int Hin, int Win, int Hout, int Wout)
{
  int idx = blockIdx.x * blockDim.x + threadIdx.x;
  int total = NC * Hout * Wout;
  if (idx >= total) return;
  int x  = idx % Wout;
  int t  = idx / Wout;
  int y  = t % Hout;
  int nc = t / Hout;
  const float* ib = in + (size_t)nc * Hin * Win;
  int y2 = 2 * y, x2 = 2 * x;
  float a = ib[(size_t)y2 * Win + x2];
  float b = ib[(size_t)y2 * Win + x2 + 1];
  float c = ib[(size_t)(y2 + 1) * Win + x2];
  float d = ib[(size_t)(y2 + 1) * Win + x2 + 1];
  out[idx] = fmaxf(fmaxf(a, b), fmaxf(c, d));
}

// ---------------- head split: sigmoid confidences + diffs ----------------
__global__ void head_split_kernel(const float* __restrict__ bb,
                                  float* __restrict__ conf, float* __restrict__ diffs)
{
  int idx = blockIdx.x * blockDim.x + threadIdx.x;
  int total = NBATCH * 45 * FEAT_HW;
  if (idx >= total) return;
  int p = idx % FEAT_HW;
  int c = (idx / FEAT_HW) % 45;
  int b = idx / (45 * FEAT_HW);
  float v = bb[idx];
  int gI = c / 9, k = c - gI * 9;
  if (gI == 0)
    conf[((size_t)b * 9 + k) * FEAT_HW + p] = 1.0f / (1.0f + __expf(-v));
  else
    diffs[(((size_t)b * 4 + (gI - 1)) * 9 + k) * FEAT_HW + p] = v;
}

// ---------------- ROI bilinear crop: (4,256,128,28,28) ----------------
__global__ void regions_kernel(const float* __restrict__ feats,
                               const int* __restrict__ boxes,
                               float* __restrict__ out)
{
  long idx = (long)blockIdx.x * blockDim.x + threadIdx.x;
  const long total = (long)NBATCH * NREG * 128 * RXY * RXY;
  if (idx >= total) return;
  int ox = idx % RXY; long t = idx / RXY;
  int oy = t % RXY;  t /= RXY;
  int c  = t % 128;  t /= 128;
  int r  = t % NREG; t /= NREG;
  int b  = (int)t;

  int ly = boxes[(b * 4 + 0) * NREG + r];
  int lx = boxes[(b * 4 + 1) * NREG + r];
  int hy = boxes[(b * 4 + 2) * NREG + r];
  int hx = boxes[(b * 4 + 3) * NREG + r];

  float Ly = (float)(hy - ly + 1);
  float gy = ((float)oy + 0.5f) * Ly / (float)RXY - 0.5f;
  gy = fminf(fmaxf(gy, 0.0f), Ly - 1.0f) + (float)ly;
  int   y0 = (int)floorf(gy);
  float fy = gy - (float)y0;
  y0 = min(y0, hy);
  int y1 = min(y0 + 1, hy);

  float Lx = (float)(hx - lx + 1);
  float gx = ((float)ox + 0.5f) * Lx / (float)RXY - 0.5f;
  gx = fminf(fmaxf(gx, 0.0f), Lx - 1.0f) + (float)lx;
  int   x0 = (int)floorf(gx);
  float fx = gx - (float)x0;
  x0 = min(x0, hx);
  int x1 = min(x0 + 1, hx);

  const float* f = feats + ((size_t)b * 128 + c) * FEAT_HW;
  float r0 = f[y0 * WP + x0] * (1.0f - fx) + f[y0 * WP + x1] * fx;
  float r1 = f[y1 * WP + x0] * (1.0f - fx) + f[y1 * WP + x1] * fx;
  out[idx] = r0 * (1.0f - fy) + r1 * fy;
}

// ---------------- IoU vs anchors, with .at[best].max(0.6) fixup ----------
__device__ inline float iou_one(int a, float gx1, float gy1, float gx2, float gy2,
                                float garea)
{
  int k   = a / FEAT_HW;
  int rem = a - k * FEAT_HW;
  int i   = rem / WP;
  int j   = rem - i * WP;
  float cy = ((float)i + 0.5f) * (1.0f / (float)HP);
  float cx = ((float)j + 0.5f) * (1.0f / (float)WP);
  const float S[3] = {0.15f, 0.45f, 0.75f};
  const float Q[3] = {0.70710678118654752f, 1.0f, 1.41421356237309515f};
  int si = k / 3, ri = k - si * 3;
  float w = S[si] * Q[ri], h = S[si] / Q[ri];
  float ax1 = cx - 0.5f * w, ay1 = cy - 0.5f * h;
  float ax2 = cx + 0.5f * w, ay2 = cy + 0.5f * h;
  float ix1 = fmaxf(gx1, ax1), iy1 = fmaxf(gy1, ay1);
  float ix2 = fminf(gx2, ax2), iy2 = fminf(gy2, ay2);
  float inter = fmaxf(ix2 - ix1, 0.0f) * fmaxf(iy2 - iy1, 0.0f);
  float aan = (ax2 - ax1) * (ay2 - ay1);
  return inter / (garea + aan - inter + 1e-9f);
}

__global__ void iou_kernel(const float* __restrict__ gt,
                           float* __restrict__ omax, float* __restrict__ oarg)
{
  const int b   = blockIdx.x;
  const int tid = threadIdx.x;
  __shared__ float sv[256];
  __shared__ int   si[256];
  __shared__ int   best[4];

  float gx1[4], gy1[4], gx2[4], gy2[4], ga[4];
  #pragma unroll
  for (int g = 0; g < 4; ++g) {
    gx1[g] = gt[(b * 4 + 0) * 4 + g];
    gy1[g] = gt[(b * 4 + 1) * 4 + g];
    gx2[g] = gt[(b * 4 + 2) * 4 + g];
    gy2[g] = gt[(b * 4 + 3) * 4 + g];
    ga[g]  = (gx2[g] - gx1[g]) * (gy2[g] - gy1[g]);
  }

  // Stage 1: per-GT argmax over anchors (first-occurrence ties)
  for (int g = 0; g < 4; ++g) {
    float bv = -1.0f; int bi = 0;
    for (int a = tid; a < NANCH; a += 256) {
      float v = iou_one(a, gx1[g], gy1[g], gx2[g], gy2[g], ga[g]);
      if (v > bv) { bv = v; bi = a; }
    }
    sv[tid] = bv; si[tid] = bi;
    __syncthreads();
    for (int st = 128; st > 0; st >>= 1) {
      if (tid < st) {
        if (sv[tid + st] > sv[tid] ||
            (sv[tid + st] == sv[tid] && si[tid + st] < si[tid])) {
          sv[tid] = sv[tid + st]; si[tid] = si[tid + st];
        }
      }
      __syncthreads();
    }
    if (tid == 0) best[g] = si[0];
    __syncthreads();
  }

  // Stage 2: per-anchor max/argmax over GT axis after 0.6 fixup
  for (int a = tid; a < NANCH; a += 256) {
    float m = -1e30f; int arg = 0;
    #pragma unroll
    for (int g = 0; g < 4; ++g) {
      float v = iou_one(a, gx1[g], gy1[g], gx2[g], gy2[g], ga[g]);
      if (a == best[g]) v = fmaxf(v, 0.6f);
      if (v > m) { m = v; arg = g; }
    }
    omax[(size_t)b * NANCH + a] = m;
    oarg[(size_t)b * NANCH + a] = (float)arg;
  }
}

// =======================================================================
extern "C" void kernel_launch(void* const* d_in, const int* in_sizes, int n_in,
                              void* d_out, int out_size, void* d_ws, size_t ws_size,
                              hipStream_t stream)
{
  const float* x     = (const float*)d_in[0];
  const int*   boxes = (const int*)d_in[1];
  const float* gt    = (const float*)d_in[2];
  const float* fw0 = (const float*)d_in[3];  const float* fb0 = (const float*)d_in[4];
  const float* fw1 = (const float*)d_in[5];  const float* fb1 = (const float*)d_in[6];
  const float* fw2 = (const float*)d_in[7];  const float* fb2 = (const float*)d_in[8];
  const float* fw3 = (const float*)d_in[9];  const float* fb3 = (const float*)d_in[10];
  const float* fw4 = (const float*)d_in[11]; const float* fb4 = (const float*)d_in[12];
  const float* fw5 = (const float*)d_in[13]; const float* fb5 = (const float*)d_in[14];
  const float* fw6 = (const float*)d_in[15]; const float* fb6 = (const float*)d_in[16];
  const float* fw7 = (const float*)d_in[17]; const float* fb7 = (const float*)d_in[18];
  const float* rw0 = (const float*)d_in[19]; const float* rb0 = (const float*)d_in[20];
  const float* rw1 = (const float*)d_in[21]; const float* rb1 = (const float*)d_in[22];

  // Output layout (floats), reference return order:
  float* out     = (float*)d_out;
  float* feats   = out;                            // 4*128*22*22   = 247808
  float* conf    = feats + 247808;                 // 4*9*22*22     = 17424
  float* diffs   = conf + 17424;                   // 4*4*9*22*22   = 69696
  float* regions = diffs + 69696;                  // 4*256*128*784 = 102760448
  float* ioumax  = regions + 102760448;            // 17424
  float* iouarg  = ioumax + 17424;                 // 17424

  // Workspace: f32 intermediates (~14.7MB) + packed f16 A/B (~16.5MB)
  float* ws  = (float*)d_ws;
  float* h0  = ws;  ws += 4 * 16 * 110 * 110;
  float* h1  = ws;  ws += 4 * 16 * 110 * 110;
  float* p1  = ws;  ws += 4 * 16 * 55 * 55;
  float* h2  = ws;  ws += 4 * 32 * 53 * 53;
  float* h3  = ws;  ws += 4 * 32 * 53 * 53;
  float* p2  = ws;  ws += 4 * 32 * 26 * 26;
  float* h4  = ws;  ws += 4 * 64 * 24 * 24;
  float* h5  = ws;  ws += 4 * 64 * 24 * 24;
  float* h6  = ws;  ws += 4 * 128 * 22 * 22;
  float* r0b = ws;  ws += 4 * 256 * 22 * 22;
  float* bbf = ws;  ws += 4 * 45 * 22 * 22;
  _Float16* pw = (_Float16*)ws;                    // packed weights (<=0.59MB)
  _Float16* pb = pw + 524288;                      // packed im2col (<=15.5MB)

  auto conv = [&](const float* in, const float* w, const float* bias, float* o,
                  int Cin, int Cout, int Hin, int Win, int pad, int ks, int relu) {
    int Hout = Hin + 2 * pad - (ks - 1), Wout = Win + 2 * pad - (ks - 1);
    int HW = Hout * Wout;
    int Ktot = Cin * ks * ks;
    int Ksteps = (Ktot + 31) / 32;
    int nM = (Cout + 15) / 16;
    int nN = (HW + 15) / 16;
    int wtotal = nM * Ksteps * 512;
    pack_weights_kernel<<<(wtotal + 255) / 256, 256, 0, stream>>>(
        w, pw, Cout, Ktot, nM, Ksteps);
    int btotal = NBATCH * nN * Ksteps * 512;
    pack_im2col_kernel<<<(btotal + 255) / 256, 256, 0, stream>>>(
        in, pb, Cin, Hin, Win, Hout, Wout, pad, ks, Ktot, Ksteps, nN);
    int nTiles = NBATCH * nN * nM;
    gemm16_wmma_kernel<<<(nTiles + 3) / 4, 128, 0, stream>>>(
        pw, pb, bias, o, Cout, nM, HW, nN, Ksteps, relu, nTiles);
  };
  auto pool = [&](const float* in, float* o, int NC, int Hin, int Win) {
    int Hout = Hin / 2, Wout = Win / 2;
    int total = NC * Hout * Wout;
    maxpool2_kernel<<<(total + 255) / 256, 256, 0, stream>>>(in, o, NC, Hin, Win, Hout, Wout);
  };

  // Backbone
  conv(x,  fw0, fb0, h0, 1,   16, 112, 112, 0, 3, 1);
  conv(h0, fw1, fb1, h1, 16,  16, 110, 110, 1, 3, 1);
  pool(h1, p1, 4 * 16, 110, 110);
  conv(p1, fw2, fb2, h2, 16,  32, 55, 55, 0, 3, 1);
  conv(h2, fw3, fb3, h3, 32,  32, 53, 53, 1, 3, 1);
  pool(h3, p2, 4 * 32, 53, 53);
  conv(p2, fw4, fb4, h4, 32,  64, 26, 26, 0, 3, 1);
  conv(h4, fw5, fb5, h5, 64,  64, 24, 24, 1, 3, 1);
  conv(h5, fw6, fb6, h6, 64, 128, 24, 24, 0, 3, 1);
  conv(h6, fw7, fb7, feats, 128, 128, 22, 22, 1, 3, 1);

  // Head (1x1 conv also on WMMA path, M padded 45 -> 48)
  conv(feats, rw0, rb0, r0b, 128, 256, 22, 22, 1, 3, 1);
  conv(r0b,   rw1, rb1, bbf, 256, 45,  22, 22, 0, 1, 0);
  {
    int total = NBATCH * 45 * FEAT_HW;
    head_split_kernel<<<(total + 255) / 256, 256, 0, stream>>>(bbf, conf, diffs);
  }

  // ROI crops (memory-bandwidth dominated: ~411 MB of stores)
  {
    long total = (long)NBATCH * NREG * 128 * RXY * RXY;
    int blocks = (int)((total + 255) / 256);
    regions_kernel<<<blocks, 256, 0, stream>>>(feats, boxes, regions);
  }

  // IoU max / argmax
  iou_kernel<<<NBATCH, 256, 0, stream>>>(gt, ioumax, iouarg);
}
